// CrossAttention_37598143709780
// MI455X (gfx1250) — matrix operand order repaired and checked by reading, not compiled, not run on hardware
//
#include <hip/hip_runtime.h>

// ---------------------------------------------------------------------------
// CDNA5 (gfx1250) WMMA cross-attention:
//   B=2, T=8, N=256, C=512, H=8, T_CTX=4 -> 1024 q / 1024 k per batch, HD=64
// Pipeline: [cvt x2][transpose-cvt x3][GEMM Q][GEMM KV][flash attn][GEMM proj]
// All matmuls on v_wmma_f32_16x16x32_f16; GEMM A-tiles staged with
// global_load_async_to_lds_b128 (ASYNCcnt), B-frags from pre-transposed f16 W^T.
// ---------------------------------------------------------------------------

typedef _Float16 v4h  __attribute__((ext_vector_type(4)));
typedef _Float16 v8h  __attribute__((ext_vector_type(8)));
typedef _Float16 v16h __attribute__((ext_vector_type(16)));
typedef float    v8f  __attribute__((ext_vector_type(8)));

#define C_DIM 512

__device__ __forceinline__ v8h ld8(const _Float16* p) { return *(const v8h*)p; }

__device__ __forceinline__ v16h cat16(v8h a, v8h b) {
  return __builtin_shufflevector(a, b, 0,1,2,3,4,5,6,7,8,9,10,11,12,13,14,15);
}

__device__ __forceinline__ v8f wmma_f16(v16h a, v16h b, v8f c) {
  // (neg_a, A, neg_b, B, c_mod, C, reuse_a, reuse_b)
  return __builtin_amdgcn_wmma_f32_16x16x32_f16(false, a, false, b, (short)0, c,
                                                false, false);
}

// ---------------------------------------------------------------------------
// Prep: f32 -> f16 elementwise (x_pred / x_ctx)
// ---------------------------------------------------------------------------
__global__ __launch_bounds__(256) void cvt_f16_kernel(
    const float* __restrict__ src, _Float16* __restrict__ dst, int n4)
{
  const int i = blockIdx.x * blockDim.x + threadIdx.x;
  if (i < n4) {
    const float4 f = ((const float4*)src)[i];
    v4h h = { (_Float16)f.x, (_Float16)f.y, (_Float16)f.z, (_Float16)f.w };
    *(v4h*)&dst[4 * i] = h;
  }
}

// ---------------------------------------------------------------------------
// Prep: W f32 [512][ncols] -> WT f16 [ncols][512]
// ---------------------------------------------------------------------------
__global__ __launch_bounds__(256) void transpose_cvt_kernel(
    const float* __restrict__ W, _Float16* __restrict__ WT, int ncols)
{
  const int c  = blockIdx.x * blockDim.x + threadIdx.x;   // column of W
  const int k0 = blockIdx.y * 8;                          // row group of W
  if (c < ncols) {
    #pragma unroll
    for (int r = 0; r < 8; r++)
      WT[(size_t)c * C_DIM + k0 + r] = (_Float16)W[(size_t)(k0 + r) * ncols + c];
  }
}

// ---------------------------------------------------------------------------
// Tiled GEMM, 128x128 block tile, K-step 32, 8 waves (wave w -> rows 16w..16w+15).
// A (f16, row-major 512) staged global->LDS via async b128; B-frags read
// directly from f16 W^T (K-contiguous per output column).
// MODE 0: Q-proj  -> Qh[bh][i][d] f16, scaled 0.125
// MODE 1: KV-proj -> Kh[bh][key][d], Vt[bh][d][key] f16
// MODE 2: O-proj (+bias) -> fp32 d_out
// ---------------------------------------------------------------------------
template<int MODE>
__global__ __launch_bounds__(256) void gemm_wmma_kernel(
    const _Float16* __restrict__ A16, const _Float16* __restrict__ WT,
    _Float16* __restrict__ Qh, _Float16* __restrict__ Kh,
    _Float16* __restrict__ Vt, float* __restrict__ OutF,
    const float* __restrict__ bias)
{
  constexpr int BM = 128, BK = 32, LDT = BK + 8;
  __shared__ __align__(16) _Float16 Asub[BM][LDT];

  const int tid  = threadIdx.x;
  const int wave = tid >> 5, lane = tid & 31;
  const int g = lane >> 4, ln = lane & 15;
  const int row0 = blockIdx.x * BM;
  const int col0 = blockIdx.y * 128;

  v8f acc[8];
  #pragma unroll
  for (int n = 0; n < 8; n++) acc[n] = {};

  // staging assignment: thread copies 32 bytes of one A row per k-step
  const int sr  = tid >> 1;
  const int sc0 = (tid & 1) * 16;
  const unsigned ldsoff = (unsigned)(size_t)&Asub[sr][sc0];
  const _Float16* Ag = A16 + (size_t)(row0 + sr) * C_DIM + sc0;

  for (int kk = 0; kk < C_DIM; kk += BK) {
    __syncthreads();
    {
      const unsigned long long ga = (unsigned long long)(size_t)(Ag + kk);
      // CDNA5 async global->LDS copy (ASYNCcnt); offset applies to both sides
      asm volatile("global_load_async_to_lds_b128 %0, %1, off"
                   :: "v"(ldsoff), "v"(ga) : "memory");
      asm volatile("global_load_async_to_lds_b128 %0, %1, off offset:16"
                   :: "v"(ldsoff), "v"(ga) : "memory");
      asm volatile("s_wait_asynccnt 0" ::: "memory");
    }
    __syncthreads();

    // A-frag (16x32): lane row = lane&15; K = 16g..16g+15
    const int arow = wave * 16 + ln;
    const v16h af = cat16(ld8(&Asub[arow][16 * g]), ld8(&Asub[arow][16 * g + 8]));
    #pragma unroll
    for (int n = 0; n < 8; n++) {
      // B-frag (32x16): lane col = lane&15; halves K = 16g..16g+15 from W^T
      const _Float16* wp = WT + (size_t)(col0 + 16 * n + ln) * C_DIM + kk + 16 * g;
      const v16h bf = cat16(ld8(wp), ld8(wp + 8));
      acc[n] = wmma_f16(af, bf, acc[n]);
    }
  }

  // Epilogue. C/D layout: reg r holds row 8g+r, col = lane&15.
  #pragma unroll
  for (int n = 0; n < 8; n++) {
    #pragma unroll
    for (int r = 0; r < 8; r++) {
      const int row = row0 + wave * 16 + 8 * g + r;
      const int col = col0 + 16 * n + ln;
      const float v = acc[n][r];
      if constexpr (MODE == 0) {
        const int b = row >> 10, i = row & 1023;
        const int h = col >> 6,  d = col & 63;
        Qh[((size_t)((b * 8 + h) * 1024 + i)) * 64 + d] = (_Float16)(v * 0.125f);
      } else if constexpr (MODE == 1) {
        const int b = row >> 10, i = row & 1023;
        const int kv = col >> 9, h = (col >> 6) & 7, d = col & 63;
        if (kv == 0)
          Kh[((size_t)((b * 8 + h) * 1024 + i)) * 64 + d] = (_Float16)v;
        else
          Vt[((size_t)((b * 8 + h) * 64 + d)) * 1024 + i] = (_Float16)v;
      } else {
        OutF[(size_t)row * C_DIM + col] = v + bias[col];
      }
    }
  }
}

// ---------------------------------------------------------------------------
// Flash attention: block = (b,h, 128-query tile); wave owns 16 queries.
// Per 32-key step: 4 WMMA for S, online softmax, P->LDS repack, 4 WMMA for O.
// ---------------------------------------------------------------------------
__global__ __launch_bounds__(256) void attn_wmma_kernel(
    const _Float16* __restrict__ Qh, const _Float16* __restrict__ Kh,
    const _Float16* __restrict__ Vt, _Float16* __restrict__ Oattn)
{
  __shared__ __align__(16) _Float16 Plds[8][16][40];

  const int tid  = threadIdx.x;
  const int wave = tid >> 5, lane = tid & 31;
  const int g = lane >> 4, ln = lane & 15;
  const int bh = blockIdx.x >> 3;
  const int qt = blockIdx.x & 7;
  const int q0 = qt * 128 + wave * 16;

  // Q A-fragments for d=0..31 and d=32..63 (rows = q0+ln, pre-scaled by HD^-0.5)
  const _Float16* Qrow = Qh + ((size_t)bh * 1024 + q0 + ln) * 64;
  const v16h qf0 = cat16(ld8(Qrow + 16 * g),      ld8(Qrow + 16 * g + 8));
  const v16h qf1 = cat16(ld8(Qrow + 32 + 16 * g), ld8(Qrow + 40 + 16 * g));

  const _Float16* Kb = Kh + (size_t)bh * 1024 * 64;   // [key][d]
  const _Float16* Vb = Vt + (size_t)bh * 64 * 1024;   // [d][key]

  v8f o[4];
  #pragma unroll
  for (int dt = 0; dt < 4; dt++) o[dt] = {};
  float m[8], l[8];
  #pragma unroll
  for (int r = 0; r < 8; r++) { m[r] = -1e30f; l[r] = 0.f; }

  for (int j0 = 0; j0 < 1024; j0 += 32) {
    // ---- S = Q K^T for 2 x 16-key tiles (B-frag: lane key = j+ln, halves d) ----
    v8f s0 = {}, s1 = {};
    const _Float16* k0p = Kb + (size_t)(j0 + ln) * 64;
    const _Float16* k1p = k0p + 16 * 64;
    if (j0 + 32 < 1024)
      __builtin_prefetch(k0p + 32 * 64, 0, 3);   // -> global_prefetch_b8
    s0 = wmma_f16(qf0, cat16(ld8(k0p + 16 * g),      ld8(k0p + 16 * g + 8)), s0);
    s0 = wmma_f16(qf1, cat16(ld8(k0p + 32 + 16 * g), ld8(k0p + 40 + 16 * g)), s0);
    s1 = wmma_f16(qf0, cat16(ld8(k1p + 16 * g),      ld8(k1p + 16 * g + 8)), s1);
    s1 = wmma_f16(qf1, cat16(ld8(k1p + 32 + 16 * g), ld8(k1p + 40 + 16 * g)), s1);

    // ---- online softmax over this 32-key slab (rows live in 16-lane groups) ----
    float mt[8], p0[8], p1[8], rs[8], sc[8];
    #pragma unroll
    for (int r = 0; r < 8; r++) mt[r] = fmaxf(s0[r], s1[r]);
    #pragma unroll
    for (int d = 1; d < 16; d <<= 1) {
      #pragma unroll
      for (int r = 0; r < 8; r++) mt[r] = fmaxf(mt[r], __shfl_xor(mt[r], d));
    }
    #pragma unroll
    for (int r = 0; r < 8; r++) {
      const float mn = fmaxf(m[r], mt[r]);
      sc[r] = __expf(m[r] - mn);
      p0[r] = __expf(s0[r] - mn);
      p1[r] = __expf(s1[r] - mn);
      rs[r] = p0[r] + p1[r];
      m[r]  = mn;
    }
    #pragma unroll
    for (int d = 1; d < 16; d <<= 1) {
      #pragma unroll
      for (int r = 0; r < 8; r++) rs[r] += __shfl_xor(rs[r], d);
    }
    #pragma unroll
    for (int r = 0; r < 8; r++) l[r] = l[r] * sc[r] + rs[r];
    #pragma unroll
    for (int dt = 0; dt < 4; dt++) {
      #pragma unroll
      for (int r = 0; r < 8; r++) o[dt][r] *= sc[r];
    }

    // ---- repack P (C/D layout) -> A-fragment layout via per-wave LDS ----
    #pragma unroll
    for (int r = 0; r < 8; r++) {
      Plds[wave][8 * g + r][ln]      = (_Float16)p0[r];
      Plds[wave][8 * g + r][16 + ln] = (_Float16)p1[r];
    }
    const v16h pf = cat16(ld8(&Plds[wave][ln][16 * g]),
                          ld8(&Plds[wave][ln][16 * g + 8]));

    // ---- O += P x V ; B-frag from Vt: lane col d = 16dt+ln, halves = keys ----
    #pragma unroll
    for (int dt = 0; dt < 4; dt++) {
      const _Float16* vp = Vb + (size_t)(16 * dt + ln) * 1024 + j0 + 16 * g;
      o[dt] = wmma_f16(pf, cat16(ld8(vp), ld8(vp + 8)), o[dt]);
    }
  }

  // ---- normalize, store in GEMM-row layout [row=b*1024+i][col=h*64+d] f16 ----
  const int b = bh >> 3, h = bh & 7;
  #pragma unroll
  for (int dt = 0; dt < 4; dt++) {
    #pragma unroll
    for (int r = 0; r < 8; r++) {
      const int row = b * 1024 + q0 + 8 * g + r;
      const int col = h * 64 + dt * 16 + ln;
      Oattn[(size_t)row * C_DIM + col] = (_Float16)(o[dt][r] / l[r]);
    }
  }
}

// ---------------------------------------------------------------------------
extern "C" void kernel_launch(void* const* d_in, const int* in_sizes, int n_in,
                              void* d_out, int out_size, void* d_ws, size_t ws_size,
                              hipStream_t stream)
{
  (void)in_sizes; (void)n_in; (void)out_size; (void)ws_size;
  const float* x_pred = (const float*)d_in[0];
  const float* x_ctx  = (const float*)d_in[1];
  // d_in[2] = ctx_mask: structurally fixed (first T_CTX slots are context)
  const float* Wq     = (const float*)d_in[3];
  const float* Wkv    = (const float*)d_in[4];
  const float* Wproj  = (const float*)d_in[5];
  const float* bproj  = (const float*)d_in[6];

  char* ws = (char*)d_ws;
  const size_t MB = 1024 * 1024;
  _Float16* Qh     = (_Float16*)(ws);                 // 2 MB  [16][1024][64]
  _Float16* Kh     = (_Float16*)(ws + 2 * MB);        // 2 MB  [16][1024][64]
  _Float16* Vt     = (_Float16*)(ws + 4 * MB);        // 2 MB  [16][64][1024]
  _Float16* Oattn  = (_Float16*)(ws + 6 * MB);        // 2 MB  [2048][512]
  _Float16* Xp16   = (_Float16*)(ws + 8 * MB);        // 2 MB  [2048][512]
  _Float16* Xc16   = (_Float16*)(ws + 10 * MB);       // 2 MB  [2048][512]
  _Float16* WqT    = (_Float16*)(ws + 12 * MB);       // 512K  [512][512]
  _Float16* WkvT   = (_Float16*)(ws + 12 * MB + 512 * 1024);   // 1 MB [1024][512]
  _Float16* WprojT = (_Float16*)(ws + 13 * MB + 512 * 1024);   // 512K [512][512]
  float* out = (float*)d_out;

  const int n4 = (2048 * C_DIM) / 4;
  cvt_f16_kernel<<<n4 / 256, 256, 0, stream>>>(x_pred, Xp16, n4);
  cvt_f16_kernel<<<n4 / 256, 256, 0, stream>>>(x_ctx,  Xc16, n4);
  transpose_cvt_kernel<<<dim3(2, 64), 256, 0, stream>>>(Wq,    WqT,    512);
  transpose_cvt_kernel<<<dim3(4, 64), 256, 0, stream>>>(Wkv,   WkvT,   1024);
  transpose_cvt_kernel<<<dim3(2, 64), 256, 0, stream>>>(Wproj, WprojT, 512);

  gemm_wmma_kernel<0><<<dim3(16, 4), 256, 0, stream>>>(
      Xp16, WqT, Qh, nullptr, nullptr, nullptr, nullptr);
  gemm_wmma_kernel<1><<<dim3(16, 8), 256, 0, stream>>>(
      Xc16, WkvT, nullptr, Kh, Vt, nullptr, nullptr);
  attn_wmma_kernel<<<128, 256, 0, stream>>>(Qh, Kh, Vt, Oattn);
  gemm_wmma_kernel<2><<<dim3(16, 4), 256, 0, stream>>>(
      Oattn, WprojT, nullptr, nullptr, nullptr, out, bproj);
}
